// RGATEncoder_10325101379598
// MI455X (gfx1250) — matrix operand order repaired and checked by reading, not compiled, not run on hardware
//
#include <hip/hip_runtime.h>
#include <hip/hip_bf16.h>

// ---------------------------------------------------------------------------
// RGAT encoder forward for MI455X (gfx1250, wave32).
//   - Dense projections x@W via v_wmma_f32_16x16x32_bf16 (bf16 in, f32 acc).
//   - Edge softmax / aggregation via L2-resident atomics (all node arrays are
//     25.6MB -> fit the 192MB L2; atomic max/add resolve at L2).
// ---------------------------------------------------------------------------

typedef __attribute__((ext_vector_type(16))) __bf16 v16bf;
typedef __attribute__((ext_vector_type(8)))  float  v8f;

#define NDIM 128   // feature dim
#define NHEAD 4    // heads (head size 32)

__device__ __forceinline__ float lrelu(float x, float s) { return x > 0.f ? x : x * s; }

__device__ __forceinline__ unsigned short f32_to_bf16(float f) {
  unsigned u = __float_as_uint(f);
  unsigned r = 0x7FFFu + ((u >> 16) & 1u);   // round-to-nearest-even
  return (unsigned short)((u + r) >> 16);
}

// Order-preserving float -> uint encoding so atomicMax(u32) == float max.
__device__ __forceinline__ unsigned enc_ord(float f) {
  unsigned u = __float_as_uint(f);
  return (u & 0x80000000u) ? ~u : (u | 0x80000000u);
}
__device__ __forceinline__ float dec_ord(unsigned u) {
  return (u & 0x80000000u) ? __uint_as_float(u & 0x7FFFFFFFu) : __uint_as_float(~u);
}

// --------------------------- LayerNorm (wave/node) -------------------------
__global__ void __launch_bounds__(256)
ln_kernel(const float* __restrict__ x, const float* __restrict__ g,
          const float* __restrict__ b, float* __restrict__ y, int n) {
  int node = (int)(blockIdx.x * 8u) + ((int)threadIdx.x >> 5);
  int lane = threadIdx.x & 31;
  if (node >= n) return;
  const float4 v = *(const float4*)(x + (size_t)node * NDIM + lane * 4);
  float s = v.x + v.y + v.z + v.w;
  #pragma unroll
  for (int m = 16; m >= 1; m >>= 1) s += __shfl_xor(s, m, 32);
  float mu = s * (1.f / NDIM);
  float dx = v.x - mu, dy = v.y - mu, dz = v.z - mu, dw = v.w - mu;
  float q = dx * dx + dy * dy + dz * dz + dw * dw;
  #pragma unroll
  for (int m = 16; m >= 1; m >>= 1) q += __shfl_xor(q, m, 32);
  float inv = rsqrtf(q * (1.f / NDIM) + 1e-5f);
  const float4 gv = *(const float4*)(g + lane * 4);
  const float4 bv = *(const float4*)(b + lane * 4);
  float4 o;
  o.x = dx * inv * gv.x + bv.x;  o.y = dy * inv * gv.y + bv.y;
  o.z = dz * inv * gv.z + bv.z;  o.w = dw * inv * gv.w + bv.w;
  *(float4*)(y + (size_t)node * NDIM + lane * 4) = o;
}

// ------------------ Weight convert: W[k][n] f32 -> Wt[n][k] bf16 -----------
__global__ void __launch_bounds__(256)
wconv_kernel(const float* __restrict__ W, unsigned short* __restrict__ Wt) {
  int i = blockIdx.x * 256 + threadIdx.x;           // 128*128 elements
  int k = i >> 7, ncol = i & 127;
  Wt[ncol * NDIM + k] = f32_to_bf16(W[k * NDIM + ncol]);
}

// ------------------------- GEMM: F = Xn @ W (WMMA bf16) --------------------
// Block: 256 thr = 8 waves; block owns 16 rows, wave w owns cols [16w,16w+16).
// K = 128 -> 4 x v_wmma_f32_16x16x32_bf16 per wave.
union Frag { v16bf v; unsigned u[8]; };

__global__ void __launch_bounds__(256)
gemm_bf16_kernel(const float* __restrict__ X, const unsigned short* __restrict__ Wt,
                 float* __restrict__ F, int n) {
  __shared__ __attribute__((aligned(16))) unsigned short lds_a[16 * NDIM];
  const int row_base = blockIdx.x * 16;
  const int tid = threadIdx.x;

  // Stage A tile (16x128) into LDS as bf16, coalesced, zero-padded past n.
  #pragma unroll
  for (int i = 0; i < 8; ++i) {
    int idx = i * 256 + tid;                 // 0..2047
    int r = idx >> 7, c = idx & 127;
    float v = (row_base + r < n) ? X[(size_t)(row_base + r) * NDIM + c] : 0.f;
    lds_a[idx] = f32_to_bf16(v);
  }
  __syncthreads();

  const int wave = tid >> 5, lane = tid & 31;
  const int col_base = wave * 16;
  const int mrow = lane & 15;                // A row / C column index
  const int half = lane >> 4;                // lane half selects K sub-range
  v8f acc = {};

  #pragma unroll
  for (int kc = 0; kc < 4; ++kc) {
    Frag a, b;
    // A fragment (16x32 bf16): VGPR0-3 -> K = kc*32 + half*8 + {0..7},
    //                          VGPR4-7 -> K = kc*32 + 16 + half*8 + {0..7}
    const unsigned short* arow = lds_a + mrow * NDIM;
    int ka = kc * 32 + half * 8;
    #pragma unroll
    for (int v = 0; v < 4; ++v) a.u[v]     = *(const unsigned*)(arow + ka + 2 * v);
    #pragma unroll
    for (int v = 0; v < 4; ++v) a.u[4 + v] = *(const unsigned*)(arow + ka + 16 + 2 * v);
    // B fragment (32x16 bf16): column = lane&15, K same as A fragment
    const unsigned short* bcol = Wt + (size_t)(col_base + mrow) * NDIM;
    int kb = kc * 32 + half * 8;
    #pragma unroll
    for (int v = 0; v < 8; ++v) b.u[v] = *(const unsigned*)(bcol + kb + (v >> 2) * 16 + 2 * (v & 3));

    acc = __builtin_amdgcn_wmma_f32_16x16x32_bf16(false, a.v, false, b.v,
                                                  (short)0, acc, false, false);
  }
  // C layout: VGPR i -> row (i + half*8), column (lane&15).
  float* fout = F + (size_t)(row_base + half * 8) * NDIM + col_base + mrow;
  if (row_base + 16 <= n) {
    // Fast path (all but the last block): 8 unguarded stores, no branches.
    #pragma unroll
    for (int i = 0; i < 8; ++i) fout[(size_t)i * NDIM] = acc[i];
  } else {
    #pragma unroll
    for (int i = 0; i < 8; ++i)
      if (row_base + half * 8 + i < n) fout[(size_t)i * NDIM] = acc[i];
  }
}

// -------- Attention coefficients: el[n,h]=sum_d f*al, er likewise ----------
__global__ void __launch_bounds__(256)
attn_coef_kernel(const float* __restrict__ f, const float* __restrict__ al,
                 const float* __restrict__ ar, float* __restrict__ el,
                 float* __restrict__ er, int n) {
  int node = (int)(blockIdx.x * 8u) + ((int)threadIdx.x >> 5);
  int lane = threadIdx.x & 31;
  if (node >= n) return;
  const float4 fv = *(const float4*)(f + (size_t)node * NDIM + lane * 4);
  const float4 av = *(const float4*)(al + lane * 4);
  const float4 rv = *(const float4*)(ar + lane * 4);
  float sl = fv.x * av.x + fv.y * av.y + fv.z * av.z + fv.w * av.w;
  float sr = fv.x * rv.x + fv.y * rv.y + fv.z * rv.z + fv.w * rv.w;
  // head = lane/8 (each lane owns 4 of the 32 per-head dims); reduce over 8 lanes
  #pragma unroll
  for (int m = 4; m >= 1; m >>= 1) { sl += __shfl_xor(sl, m, 32); sr += __shfl_xor(sr, m, 32); }
  if ((lane & 7) == 0) {
    el[node * NHEAD + (lane >> 3)] = sl;
    er[node * NHEAD + (lane >> 3)] = sr;
  }
}

// --------------------- Edge pass 1: segment max of e -----------------------
__global__ void __launch_bounds__(256)
edge_max_kernel(const int* __restrict__ src, const int* __restrict__ dst,
                const float* __restrict__ el, const float* __restrict__ er,
                unsigned* __restrict__ menc, int ne) {
  int e = blockIdx.x * 256 + threadIdx.x;
  if (e >= ne) return;
  int s = src[e], d = dst[e];
  const float4 l = *(const float4*)(el + (size_t)s * NHEAD);
  const float4 r = *(const float4*)(er + (size_t)d * NHEAD);
  float v[4] = { l.x + r.x, l.y + r.y, l.z + r.z, l.w + r.w };
  #pragma unroll
  for (int h = 0; h < NHEAD; ++h)
    atomicMax(&menc[(size_t)d * NHEAD + h], enc_ord(lrelu(v[h], 0.2f)));
}

// --------------------- Edge pass 2: segment sum of exp ---------------------
__global__ void __launch_bounds__(256)
edge_sum_kernel(const int* __restrict__ src, const int* __restrict__ dst,
                const float* __restrict__ el, const float* __restrict__ er,
                const unsigned* __restrict__ menc, float* __restrict__ ssum, int ne) {
  int e = blockIdx.x * 256 + threadIdx.x;
  if (e >= ne) return;
  int s = src[e], d = dst[e];
  const float4 l = *(const float4*)(el + (size_t)s * NHEAD);
  const float4 r = *(const float4*)(er + (size_t)d * NHEAD);
  float v[4] = { l.x + r.x, l.y + r.y, l.z + r.z, l.w + r.w };
  #pragma unroll
  for (int h = 0; h < NHEAD; ++h) {
    float t = lrelu(v[h], 0.2f);
    float m = dec_ord(menc[(size_t)d * NHEAD + h]);
    atomicAdd(&ssum[(size_t)d * NHEAD + h], __expf(t - m));
  }
}

// ------------- Edge pass 3: scatter alpha-weighted messages ----------------
// One wave per edge: 32 lanes x 4 floats = 128 message elements.
__global__ void __launch_bounds__(256)
edge_msg_kernel(const int* __restrict__ src, const int* __restrict__ dst,
                const float* __restrict__ f, const float* __restrict__ el,
                const float* __restrict__ er, const unsigned* __restrict__ menc,
                const float* __restrict__ ssum, float* __restrict__ acc, int ne) {
  int e = (int)(blockIdx.x * 8u) + ((int)threadIdx.x >> 5);
  int lane = threadIdx.x & 31;
  if (e >= ne) return;
  int s = src[e], d = dst[e];
  int h = lane >> 3;                           // head owned by this lane group
  float t = lrelu(el[(size_t)s * NHEAD + h] + er[(size_t)d * NHEAD + h], 0.2f);
  float m = dec_ord(menc[(size_t)d * NHEAD + h]);
  float den = ssum[(size_t)d * NHEAD + h];
  float alpha = __expf(t - m) / (den == 0.f ? 1.f : den);
  const float4 fv = *(const float4*)(f + (size_t)s * NDIM + lane * 4);
  float* o = acc + (size_t)d * NDIM + lane * 4;
  atomicAdd(o + 0, fv.x * alpha);
  atomicAdd(o + 1, fv.y * alpha);
  atomicAdd(o + 2, fv.z * alpha);
  atomicAdd(o + 3, fv.w * alpha);
}

// ------- Combine relations: mean of lrelu(acc_r + bias_r) (+ residual) -----
__global__ void __launch_bounds__(256)
combine_kernel(const float* __restrict__ a0, const float* __restrict__ a1,
               const float* __restrict__ b0, const float* __restrict__ b1,
               const float* __restrict__ res, float* __restrict__ out, int n) {
  int i = blockIdx.x * 256 + threadIdx.x;
  if (i >= n * NDIM) return;
  int c = i & (NDIM - 1);
  float x0 = lrelu(a0[i] + b0[c], 0.1f);
  float x1 = lrelu(a1[i] + b1[c], 0.1f);
  float o = 0.5f * (x0 + x1);
  if (res) o += res[i];
  out[i] = o;
}

// ---------------------------------------------------------------------------
extern "C" void kernel_launch(void* const* d_in, const int* in_sizes, int n_in,
                              void* d_out, int out_size, void* d_ws, size_t ws_size,
                              hipStream_t stream) {
  const float* feat   = (const float*)d_in[0];
  const float* ln0_g  = (const float*)d_in[1];
  const float* ln0_b  = (const float*)d_in[2];
  const float* ln1_g  = (const float*)d_in[3];
  const float* ln1_b  = (const float*)d_in[4];
  const float* emb_W  = (const float*)d_in[5];
  const float* emb_al = (const float*)d_in[6];
  const float* emb_ar = (const float*)d_in[7];
  const float* emb_b  = (const float*)d_in[8];
  const float* conv_W = (const float*)d_in[9];
  const float* conv_al= (const float*)d_in[10];
  const float* conv_ar= (const float*)d_in[11];
  const float* conv_b = (const float*)d_in[12];
  const int* srcs[2]  = { (const int*)d_in[13], (const int*)d_in[15] };
  const int* dsts[2]  = { (const int*)d_in[14], (const int*)d_in[16] };

  const int n  = in_sizes[0] / NDIM;     // 50000
  const int ne = in_sizes[13];           // 500000
  float* out = (float*)d_out;            // doubles as the hidden state h

  // ---- workspace carve-up (256B aligned) ----
  char* wp = (char*)d_ws;
  auto carve = [&](size_t bytes) {
    char* p = wp; wp += (bytes + 255) & ~(size_t)255; return (void*)p;
  };
  float*  xn   = (float*)carve((size_t)n * NDIM * 4);   // LN output / hn
  float*  f    = (float*)carve((size_t)n * NDIM * 4);   // projected features
  float*  acc0 = (float*)carve((size_t)n * NDIM * 4);
  float*  acc1 = (float*)carve((size_t)n * NDIM * 4);
  float*  emb  = (float*)carve((size_t)n * NDIM * 4);   // residual
  float*  el   = (float*)carve((size_t)n * NHEAD * 4);
  float*  er   = (float*)carve((size_t)n * NHEAD * 4);
  unsigned* menc = (unsigned*)carve((size_t)n * NHEAD * 4);
  float*  ssum = (float*)carve((size_t)n * NHEAD * 4);
  unsigned short* wt = (unsigned short*)carve((size_t)NDIM * NDIM * 2);

  const int nodeWaveBlocks = (n + 7) / 8;
  const int gemmBlocks     = (n + 15) / 16;
  const int edgeBlocks     = (ne + 255) / 256;
  const int edgeWaveBlocks = (ne + 7) / 8;
  const int elemBlocks     = (n * NDIM + 255) / 256;

  auto hetero = [&](const float* x_in, const float* W, const float* al,
                    const float* ar, const float* bias, const float* res,
                    float* dest) {
    for (int r = 0; r < 2; ++r) {
      float* acc = r ? acc1 : acc0;
      hipMemsetAsync(acc,  0, (size_t)n * NDIM * 4, stream);
      hipMemsetAsync(menc, 0, (size_t)n * NHEAD * 4, stream);   // enc-ordered -inf
      hipMemsetAsync(ssum, 0, (size_t)n * NHEAD * 4, stream);
      wconv_kernel<<<64, 256, 0, stream>>>(W + (size_t)r * NDIM * NDIM, wt);
      gemm_bf16_kernel<<<gemmBlocks, 256, 0, stream>>>(x_in, wt, f, n);
      attn_coef_kernel<<<nodeWaveBlocks, 256, 0, stream>>>(
          f, al + r * NDIM, ar + r * NDIM, el, er, n);
      edge_max_kernel<<<edgeBlocks, 256, 0, stream>>>(srcs[r], dsts[r], el, er, menc, ne);
      edge_sum_kernel<<<edgeBlocks, 256, 0, stream>>>(srcs[r], dsts[r], el, er, menc, ssum, ne);
      edge_msg_kernel<<<edgeWaveBlocks, 256, 0, stream>>>(
          srcs[r], dsts[r], f, el, er, menc, ssum, acc, ne);
    }
    combine_kernel<<<elemBlocks, 256, 0, stream>>>(
        acc0, acc1, bias, bias + NDIM, res, dest, n);
  };

  // x = LN(feat); emb = hetero(x, emb params); h = emb
  ln_kernel<<<nodeWaveBlocks, 256, 0, stream>>>(feat, ln0_g, ln0_b, xn, n);
  hetero(xn, emb_W, emb_al, emb_ar, emb_b, nullptr, emb);
  hipMemcpyAsync(out, emb, (size_t)n * NDIM * 4, hipMemcpyDeviceToDevice, stream);

  // 2 GAT layers with residual to emb; h lives in d_out
  for (int layer = 0; layer < 2; ++layer) {
    ln_kernel<<<nodeWaveBlocks, 256, 0, stream>>>(out, ln1_g, ln1_b, xn, n);
    hetero(xn, conv_W, conv_al, conv_ar, conv_b, emb, out);
  }
}
